// TokenEmbeder_19834158973303
// MI455X (gfx1250) — matrix operand order repaired and checked by reading, not compiled, not run on hardware
//
#include <hip/hip_runtime.h>
#include <math.h>

// ---------------------------------------------------------------------------
// Problem constants (from reference): B=64, ST=64, SD=32, E=H=512, 4H=2048
// ---------------------------------------------------------------------------
#define B_   64
#define ST_  64
#define SD_  32
#define E_   512
#define H_   512
#define G4_  2048

typedef __attribute__((ext_vector_type(16))) __bf16        v16bf;
typedef __attribute__((ext_vector_type(8)))  float         v8f;
typedef __attribute__((ext_vector_type(4)))  unsigned int  v4u;

union Frag16 { v16bf v; v4u q[2]; };

__device__ __forceinline__ unsigned short f2bf(float f) {
  unsigned int u = __float_as_uint(f);
  unsigned int r = (u + 0x7FFFu + ((u >> 16) & 1u)) >> 16;   // RNE
  return (unsigned short)r;
}

// CDNA5 hardware transcendental tanh (TRANS op). The trailing v_nop satisfies
// the ISA's required gap after a trans op before its sources or result are
// reused (compiler can't see inside the asm).
__device__ __forceinline__ float fast_tanh(float x) {
  float y;
  asm("v_tanh_f32 %0, %1\n\tv_nop" : "=v"(y) : "v"(x));
  return y;
}
// sigmoid(x) = 0.5*tanh(x/2) + 0.5  -> one TRANS op + FMA
__device__ __forceinline__ float fast_sigmoid(float x) {
  return __builtin_fmaf(0.5f, fast_tanh(0.5f * x), 0.5f);
}

// ---------------------------------------------------------------------------
// Elementwise helpers
// ---------------------------------------------------------------------------
__global__ void cast_f32_bf16_k(const float* __restrict__ src,
                                unsigned short* __restrict__ dst, int n) {
  for (int i = blockIdx.x * blockDim.x + threadIdx.x; i < n;
       i += gridDim.x * blockDim.x)
    dst[i] = f2bf(src[i]);
}

__global__ void bias_combine_k(const float* __restrict__ a,
                               const float* __restrict__ b,
                               float* __restrict__ out, int n) {
  for (int i = blockIdx.x * blockDim.x + threadIdx.x; i < n;
       i += gridDim.x * blockDim.x)
    out[i] = a[i] + b[i];
}

// gather rows of the embedding table, cast to bf16: X[r, e] = emb[ids[r], e]
__global__ void gather_embed_k(const int* __restrict__ ids,
                               const float* __restrict__ emb,
                               unsigned short* __restrict__ out, int n /*rows*E*/) {
  for (int i = blockIdx.x * blockDim.x + threadIdx.x; i < n;
       i += gridDim.x * blockDim.x) {
    int r = i >> 9;          // E_ = 512
    int e = i & (E_ - 1);
    out[i] = f2bf(emb[(size_t)ids[r] * E_ + e]);
  }
}

// ---------------------------------------------------------------------------
// Pre-gate GEMM (hoisted out of the scan):
//   G[s][b][n] = sum_k X[b*S+s, k] * Wih[n, k] + bias[n]
// One wave computes a 16x64 block. Inner K loop is 2x unrolled with explicit
// double-buffered fragments so VMEM latency overlaps the WMMA pipe.
// ---------------------------------------------------------------------------
__global__ void pregate_gemm_k(const unsigned short* __restrict__ X,
                               const unsigned short* __restrict__ W,
                               const float* __restrict__ bias,
                               float* __restrict__ G, int lgS) {
  const int lane = threadIdx.x & 31;
  const int wid  = threadIdx.x >> 5;
  const int t    = blockIdx.x * 8 + wid;   // global wave id
  const int strip = t >> 5;                // 16-row strip
  const int cg    = t & 31;                // 64-col group (2048/64)

  const int nloc = lane & 15;
  const int ao   = (lane < 16) ? 0 : 16;   // A K-offset (contiguous layout)
  const int bo   = (lane < 16) ? 0 : 16;   // B K-offset
  const unsigned short* Arow = X + (size_t)(strip * 16 + nloc) * E_;
  const unsigned short* Bbase = W + (size_t)(cg * 64 + nloc) * E_ + bo;

  auto loadA = [&](Frag16& f, int kk) {
    f.q[0] = *(const v4u*)(Arow + kk * 32 + ao);
    f.q[1] = *(const v4u*)(Arow + kk * 32 + ao + 8);
  };
  auto loadB = [&](Frag16* f, int kk) {
    const unsigned short* Bp = Bbase + kk * 32;
#pragma unroll
    for (int cc = 0; cc < 4; ++cc) {
      f[cc].q[0] = *(const v4u*)(Bp + (size_t)cc * 16 * E_);
      f[cc].q[1] = *(const v4u*)(Bp + (size_t)cc * 16 * E_ + 8);
    }
  };

  v8f acc[4] = {};
  Frag16 a0, a1, b0[4], b1[4];
  loadA(a0, 0); loadB(b0, 0);
#pragma unroll 1
  for (int kk = 0; kk < 16; kk += 2) {
    loadA(a1, kk + 1); loadB(b1, kk + 1);
    __builtin_prefetch((const void*)(Arow + kk * 32 + 512), 0, 3);
#pragma unroll
    for (int cc = 0; cc < 4; ++cc)
      acc[cc] = __builtin_amdgcn_wmma_f32_16x16x32_bf16(
          false, a0.v, false, b0[cc].v, (short)0, acc[cc], false, false);
    if (kk + 2 < 16) { loadA(a0, kk + 2); loadB(b0, kk + 2); }
#pragma unroll
    for (int cc = 0; cc < 4; ++cc)
      acc[cc] = __builtin_amdgcn_wmma_f32_16x16x32_bf16(
          false, a1.v, false, b1[cc].v, (short)0, acc[cc], false, false);
  }

  // epilogue: scatter into [s][b][n] with bias
  const int half8 = (lane < 16) ? 0 : 8;
  const int Smask = (1 << lgS) - 1;
#pragma unroll
  for (int cc = 0; cc < 4; ++cc) {
    const int col = cg * 64 + cc * 16 + nloc;
    const float bv = bias[col];
#pragma unroll
    for (int k = 0; k < 8; ++k) {
      const int row = strip * 16 + k + half8;    // row = b*S + s
      const int bb = row >> lgS;
      const int ss = row & Smask;
      G[(size_t)(ss * B_ + bb) * G4_ + col] = acc[cc][k] + bv;
    }
  }
}

// ---------------------------------------------------------------------------
// LSTM recurrence. One persistent 512-thread workgroup (16 waves) per LSTM
// (block 0: token S=64, block 1: anchor S=32, block 2: neg S=32).
// h kept bf16 in LDS, double-buffered (128 KB of the 320 KB WGP LDS).
// c kept in registers in the WMMA C/D fragment layout. Wave w owns hidden
// column tiles j in {w, w+16}; per (j, row-tile r) it computes the 4 gate
// tiles of h @ Whh^T with a software-pipelined K loop.
// ---------------------------------------------------------------------------
__global__ void lstm_scan_k(const float* __restrict__ Gtok,
                            const float* __restrict__ Ganc,
                            const float* __restrict__ Gneg,
                            const unsigned short* __restrict__ WhhT,
                            const unsigned short* __restrict__ WhhD,
                            float* __restrict__ featT, float* __restrict__ featA,
                            float* __restrict__ featN,
                            unsigned short* __restrict__ featTbf,
                            unsigned short* __restrict__ featAbf,
                            unsigned short* __restrict__ featNbf) {
  extern __shared__ unsigned short hshm[];   // 2 buffers of 64*512 bf16

  const float* G; const unsigned short* Whh;
  float* feat; unsigned short* featbf; int S;
  if (blockIdx.x == 0)      { G = Gtok; Whh = WhhT; feat = featT; featbf = featTbf; S = ST_; }
  else if (blockIdx.x == 1) { G = Ganc; Whh = WhhD; feat = featA; featbf = featAbf; S = SD_; }
  else                      { G = Gneg; Whh = WhhD; feat = featN; featbf = featNbf; S = SD_; }

  const int lane  = threadIdx.x & 31;
  const int w     = threadIdx.x >> 5;         // 0..15
  const int nloc  = lane & 15;
  const int ao    = (lane < 16) ? 0 : 16;
  const int bo    = (lane < 16) ? 0 : 16;
  const int half8 = (lane < 16) ? 0 : 8;

  // h(0) = 0
  for (int i = threadIdx.x; i < B_ * H_; i += 512) hshm[i] = 0;
  __syncthreads();

  float c[2][4][8];
#pragma unroll
  for (int jj = 0; jj < 2; ++jj)
#pragma unroll
    for (int r = 0; r < 4; ++r)
#pragma unroll
      for (int k = 0; k < 8; ++k) c[jj][r][k] = 0.f;

  for (int s = 0; s < S; ++s) {
    const unsigned short* hin  = hshm + ((s & 1) ? (B_ * H_) : 0);
    unsigned short*       hout = hshm + ((s & 1) ? 0 : (B_ * H_));
    const float* Gs = G + (size_t)s * (B_ * G4_);

#pragma unroll 1
    for (int jj = 0; jj < 2; ++jj) {
      const int j = w + jj * 16;          // hidden-col tile, 0..31
      const int colh = j * 16 + nloc;     // hidden index 0..511
      const unsigned short* Bbase = Whh + (size_t)colh * H_ + bo;
#pragma unroll 1
      for (int r = 0; r < 4; ++r) {       // batch-row tile, 0..3
        const unsigned short* Ar = hin + (size_t)(r * 16 + nloc) * H_;

        auto loadA = [&](Frag16& f, int kk) {
          f.q[0] = *(const v4u*)(Ar + kk * 32 + ao);
          f.q[1] = *(const v4u*)(Ar + kk * 32 + ao + 8);
        };
        auto loadB = [&](Frag16* f, int kk) {
          const unsigned short* Bp = Bbase + kk * 32;
#pragma unroll
          for (int q = 0; q < 4; ++q) {   // i,f,g,o gate rows (+512*H each)
            f[q].q[0] = *(const v4u*)(Bp + (size_t)q * 512 * H_);
            f[q].q[1] = *(const v4u*)(Bp + (size_t)q * 512 * H_ + 8);
          }
        };

        v8f acc[4] = {};
        Frag16 a0, a1, b0[4], b1[4];
        loadA(a0, 0); loadB(b0, 0);
#pragma unroll 1
        for (int kk = 0; kk < 16; kk += 2) {
          loadA(a1, kk + 1); loadB(b1, kk + 1);
#pragma unroll
          for (int q = 0; q < 4; ++q)
            acc[q] = __builtin_amdgcn_wmma_f32_16x16x32_bf16(
                false, a0.v, false, b0[q].v, (short)0, acc[q], false, false);
          if (kk + 2 < 16) { loadA(a0, kk + 2); loadB(b0, kk + 2); }
#pragma unroll
          for (int q = 0; q < 4; ++q)
            acc[q] = __builtin_amdgcn_wmma_f32_16x16x32_bf16(
                false, a1.v, false, b1[q].v, (short)0, acc[q], false, false);
        }

        // gate pointwise + state update (register-local, fragment layout)
#pragma unroll
        for (int k = 0; k < 8; ++k) {
          const int brow = r * 16 + k + half8;           // batch index
          const float* Grow = Gs + (size_t)brow * G4_;
          const float gi = acc[0][k] + Grow[colh];
          const float gf = acc[1][k] + Grow[512 + colh];
          const float gg = acc[2][k] + Grow[1024 + colh];
          const float go = acc[3][k] + Grow[1536 + colh];
          const float cn = fast_sigmoid(gf) * c[jj][r][k] +
                           fast_sigmoid(gi) * fast_tanh(gg);
          c[jj][r][k] = cn;
          const float hv = fast_sigmoid(go) * fast_tanh(cn);
          const unsigned short hb = f2bf(hv);
          hout[(size_t)brow * H_ + colh] = hb;
          const size_t fi = ((size_t)brow * S + s) * H_ + colh;
          feat[fi]   = hv;
          featbf[fi] = hb;
        }
      }
    }
    __syncthreads();   // step boundary: all reads of hin done, hout complete
  }
}

// ---------------------------------------------------------------------------
// Attention logits: logit[r] = wv . tanh(feat[r] @ Wa^T + ba) + bv
// One wave per 16-row strip; sweeps all 32 column tiles of H, fusing the
// tanh + wv dot into the epilogue with 16-lane shuffle reductions.
// ---------------------------------------------------------------------------
__global__ void attn_logits_k(const unsigned short* __restrict__ featbf,
                              const unsigned short* __restrict__ Wabf,
                              const float* __restrict__ ba,
                              const float* __restrict__ wv,
                              const float* __restrict__ bv,
                              float* __restrict__ logits) {
  const int lane  = threadIdx.x & 31;
  const int wid   = threadIdx.x >> 5;
  const int strip = blockIdx.x * 8 + wid;
  const int nloc  = lane & 15;
  const int ao    = (lane < 16) ? 0 : 16;
  const int bo    = (lane < 16) ? 0 : 16;
  const unsigned short* Arow = featbf + (size_t)(strip * 16 + nloc) * H_;

  float part[8] = {};
#pragma unroll 1
  for (int ct = 0; ct < 32; ++ct) {
    const int col = ct * 16 + nloc;
    const unsigned short* Brow = Wabf + (size_t)col * H_ + bo;

    auto loadA = [&](Frag16& f, int kk) {
      f.q[0] = *(const v4u*)(Arow + kk * 32 + ao);
      f.q[1] = *(const v4u*)(Arow + kk * 32 + ao + 8);
    };
    auto loadB = [&](Frag16& f, int kk) {
      f.q[0] = *(const v4u*)(Brow + kk * 32);
      f.q[1] = *(const v4u*)(Brow + kk * 32 + 8);
    };

    v8f acc = {};
    Frag16 a0, a1, b0, b1;
    loadA(a0, 0); loadB(b0, 0);
#pragma unroll 1
    for (int kk = 0; kk < 16; kk += 2) {
      loadA(a1, kk + 1); loadB(b1, kk + 1);
      __builtin_prefetch((const void*)(Brow + kk * 32 + 512), 0, 3);
      acc = __builtin_amdgcn_wmma_f32_16x16x32_bf16(
          false, a0.v, false, b0.v, (short)0, acc, false, false);
      if (kk + 2 < 16) { loadA(a0, kk + 2); loadB(b0, kk + 2); }
      acc = __builtin_amdgcn_wmma_f32_16x16x32_bf16(
          false, a1.v, false, b1.v, (short)0, acc, false, false);
    }
    const float bav = ba[col];
    const float wvv = wv[col];
#pragma unroll
    for (int k = 0; k < 8; ++k) part[k] += fast_tanh(acc[k] + bav) * wvv;
  }
  // reduce over the 16 lanes that share a row (each lane holds one N column)
#pragma unroll
  for (int k = 0; k < 8; ++k) {
    float v = part[k];
    v += __shfl_xor(v, 1, 16);
    v += __shfl_xor(v, 2, 16);
    v += __shfl_xor(v, 4, 16);
    v += __shfl_xor(v, 8, 16);
    part[k] = v;
  }
  if ((lane & 15) == 0) {
    const int rbase = strip * 16 + ((lane < 16) ? 0 : 8);
    const float bvv = bv[0];
#pragma unroll
    for (int k = 0; k < 8; ++k) logits[rbase + k] = part[k] + bvv;
  }
}

// ---------------------------------------------------------------------------
// Masked softmax + pooling + double tanh. One block per batch sample.
// ---------------------------------------------------------------------------
__global__ void attn_pool_k(const float* __restrict__ feat,
                            const float* __restrict__ logits,
                            const int* __restrict__ lens,
                            float* __restrict__ repr, int S) {
  __shared__ float wsh[64];
  __shared__ float inv_s;
  const int b = blockIdx.x;
  const int t = threadIdx.x;         // 128 threads
  const int len = lens[b];
  if (t < S) wsh[t] = (t < len) ? logits[b * S + t] : -1e9f;
  __syncthreads();
  if (t == 0) {
    float m = -1e30f;
    for (int s = 0; s < S; ++s) m = fmaxf(m, wsh[s]);
    float sum = 0.f;
    for (int s = 0; s < S; ++s) { float e = __expf(wsh[s] - m); wsh[s] = e; sum += e; }
    inv_s = 1.f / sum;
  }
  __syncthreads();
  const float inv = inv_s;
  for (int h = t; h < H_; h += 128) {
    float acc = 0.f;
    for (int s = 0; s < S; ++s) acc += wsh[s] * feat[((size_t)b * S + s) * H_ + h];
    acc *= inv;
    repr[(size_t)b * H_ + h] = fast_tanh(fast_tanh(acc));
  }
}

// ---------------------------------------------------------------------------
// Cosine margin ranking loss (mean over batch).
// ---------------------------------------------------------------------------
__global__ void loss_k(const float* __restrict__ code,
                       const float* __restrict__ anc,
                       const float* __restrict__ neg,
                       float* __restrict__ out) {
  __shared__ float sh[B_];
  const int b = threadIdx.x;   // 64 threads
  float dca = 0, dcn = 0, nc = 0, na = 0, nn = 0;
  for (int h = 0; h < H_; ++h) {
    const float cv = code[(size_t)b * H_ + h];
    const float av = anc[(size_t)b * H_ + h];
    const float nv = neg[(size_t)b * H_ + h];
    dca += cv * av; dcn += cv * nv;
    nc += cv * cv; na += av * av; nn += nv * nv;
  }
  const float eps = 1e-8f;
  const float ncn = fmaxf(sqrtf(nc), eps);
  const float cosa = dca / (ncn * fmaxf(sqrtf(na), eps));
  const float cosn = dcn / (ncn * fmaxf(sqrtf(nn), eps));
  sh[b] = fmaxf(0.6f - cosa + cosn, 1e-6f);
  __syncthreads();
  if (b == 0) {
    float s = 0.f;
    for (int i = 0; i < B_; ++i) s += sh[i];
    out[0] = s / (float)B_;
  }
}

// ---------------------------------------------------------------------------
// Host launcher
// ---------------------------------------------------------------------------
extern "C" void kernel_launch(void* const* d_in, const int* in_sizes, int n_in,
                              void* d_out, int out_size, void* d_ws, size_t ws_size,
                              hipStream_t stream) {
  (void)in_sizes; (void)n_in; (void)out_size; (void)ws_size;

  const int*   token     = (const int*)d_in[0];
  const int*   token_len = (const int*)d_in[1];
  const int*   anc_ids   = (const int*)d_in[2];
  const int*   anc_len   = (const int*)d_in[3];
  const int*   neg_ids   = (const int*)d_in[4];
  const int*   neg_len   = (const int*)d_in[5];
  const float* tok_emb   = (const float*)d_in[6];
  const float* desc_emb  = (const float*)d_in[7];
  const float* tok_Wih   = (const float*)d_in[8];
  const float* tok_Whh   = (const float*)d_in[9];
  const float* tok_bih   = (const float*)d_in[10];
  const float* tok_bhh   = (const float*)d_in[11];
  const float* d_Wih     = (const float*)d_in[12];
  const float* d_Whh     = (const float*)d_in[13];
  const float* d_bih     = (const float*)d_in[14];
  const float* d_bhh     = (const float*)d_in[15];
  const float* tok_Wa    = (const float*)d_in[16];
  const float* tok_ba    = (const float*)d_in[17];
  const float* tok_wv    = (const float*)d_in[18];
  const float* tok_bv    = (const float*)d_in[19];
  const float* d_Wa      = (const float*)d_in[20];
  const float* d_ba      = (const float*)d_in[21];
  const float* d_wv      = (const float*)d_in[22];
  const float* d_bv      = (const float*)d_in[23];

  char* base = (char*)d_ws;
  size_t off = 0;
  auto take = [&](size_t bytes) -> void* {
    void* r = base + off;
    off = (off + bytes + 255) & ~(size_t)255;
    return r;
  };

  unsigned short* Xtok   = (unsigned short*)take((size_t)B_ * ST_ * E_ * 2);
  unsigned short* Xanc   = (unsigned short*)take((size_t)B_ * SD_ * E_ * 2);
  unsigned short* Xneg   = (unsigned short*)take((size_t)B_ * SD_ * E_ * 2);
  unsigned short* WihTb  = (unsigned short*)take((size_t)G4_ * E_ * 2);
  unsigned short* WhhTb  = (unsigned short*)take((size_t)G4_ * H_ * 2);
  unsigned short* WihDb  = (unsigned short*)take((size_t)G4_ * E_ * 2);
  unsigned short* WhhDb  = (unsigned short*)take((size_t)G4_ * H_ * 2);
  unsigned short* WaTb   = (unsigned short*)take((size_t)H_ * H_ * 2);
  unsigned short* WaDb   = (unsigned short*)take((size_t)H_ * H_ * 2);
  float* biasT           = (float*)take((size_t)G4_ * 4);
  float* biasD           = (float*)take((size_t)G4_ * 4);
  float* Gtok            = (float*)take((size_t)ST_ * B_ * G4_ * 4);
  float* Ganc            = (float*)take((size_t)SD_ * B_ * G4_ * 4);
  float* Gneg            = (float*)take((size_t)SD_ * B_ * G4_ * 4);
  float* featT           = (float*)take((size_t)B_ * ST_ * H_ * 4);
  float* featA           = (float*)take((size_t)B_ * SD_ * H_ * 4);
  float* featN           = (float*)take((size_t)B_ * SD_ * H_ * 4);
  unsigned short* featTb = (unsigned short*)take((size_t)B_ * ST_ * H_ * 2);
  unsigned short* featAb = (unsigned short*)take((size_t)B_ * SD_ * H_ * 2);
  unsigned short* featNb = (unsigned short*)take((size_t)B_ * SD_ * H_ * 2);
  float* logT            = (float*)take((size_t)B_ * ST_ * 4);
  float* logA            = (float*)take((size_t)B_ * SD_ * 4);
  float* logN            = (float*)take((size_t)B_ * SD_ * 4);
  float* reprC           = (float*)take((size_t)B_ * H_ * 4);
  float* reprA           = (float*)take((size_t)B_ * H_ * 4);
  float* reprN           = (float*)take((size_t)B_ * H_ * 4);

  // 1) weight casts + bias combine
  cast_f32_bf16_k<<<512, 256, 0, stream>>>(tok_Wih, WihTb, G4_ * E_);
  cast_f32_bf16_k<<<512, 256, 0, stream>>>(tok_Whh, WhhTb, G4_ * H_);
  cast_f32_bf16_k<<<512, 256, 0, stream>>>(d_Wih,  WihDb, G4_ * E_);
  cast_f32_bf16_k<<<512, 256, 0, stream>>>(d_Whh,  WhhDb, G4_ * H_);
  cast_f32_bf16_k<<<256, 256, 0, stream>>>(tok_Wa, WaTb,  H_ * H_);
  cast_f32_bf16_k<<<256, 256, 0, stream>>>(d_Wa,   WaDb,  H_ * H_);
  bias_combine_k<<<8, 256, 0, stream>>>(tok_bih, tok_bhh, biasT, G4_);
  bias_combine_k<<<8, 256, 0, stream>>>(d_bih,   d_bhh,   biasD, G4_);

  // 2) embedding gather (fp32 -> bf16 rows)
  gather_embed_k<<<2048, 256, 0, stream>>>(token,   tok_emb,  Xtok, B_ * ST_ * E_);
  gather_embed_k<<<1024, 256, 0, stream>>>(anc_ids, desc_emb, Xanc, B_ * SD_ * E_);
  gather_embed_k<<<1024, 256, 0, stream>>>(neg_ids, desc_emb, Xneg, B_ * SD_ * E_);

  // 3) pre-gate GEMMs (hoisted X @ Wih^T + bias), WMMA bf16
  pregate_gemm_k<<<1024, 256, 0, stream>>>(Xtok, WihTb, biasT, Gtok, 6);
  pregate_gemm_k<<<512,  256, 0, stream>>>(Xanc, WihDb, biasD, Ganc, 5);
  pregate_gemm_k<<<512,  256, 0, stream>>>(Xneg, WihDb, biasD, Gneg, 5);

  // 4) the three LSTM scans: persistent workgroups, h double-buffered in LDS
  lstm_scan_k<<<3, 512, 2 * B_ * H_ * sizeof(unsigned short), stream>>>(
      Gtok, Ganc, Gneg, WhhTb, WhhDb,
      featT, featA, featN, featTb, featAb, featNb);

  // 5) attention logits via WMMA with fused tanh + wv-dot epilogue
  attn_logits_k<<<32, 256, 0, stream>>>(featTb, WaTb, tok_ba, tok_wv, tok_bv, logT);
  attn_logits_k<<<16, 256, 0, stream>>>(featAb, WaDb, d_ba,   d_wv,   d_bv,   logA);
  attn_logits_k<<<16, 256, 0, stream>>>(featNb, WaDb, d_ba,   d_wv,   d_bv,   logN);

  // 6) masked softmax + pooling + tanh(tanh(.))
  attn_pool_k<<<B_, 128, 0, stream>>>(featT, logT, token_len, reprC, ST_);
  attn_pool_k<<<B_, 128, 0, stream>>>(featA, logA, anc_len,   reprA, SD_);
  attn_pool_k<<<B_, 128, 0, stream>>>(featN, logN, neg_len,   reprN, SD_);

  // 7) cosine margin loss
  loss_k<<<1, B_, 0, stream>>>(reprC, reprA, reprN, (float*)d_out);
}
